// Block_39814346834514
// MI455X (gfx1250) — matrix operand order repaired and checked by reading, not compiled, not run on hardware
//
#include <hip/hip_runtime.h>
#include <stdint.h>

// ---------------------------------------------------------------------------
// Sizes (compile-time, from the reference)
// ---------------------------------------------------------------------------
#define HEADS   16
#define D_MODEL 1024
#define D_EXP   2048
#define D_HEAD  128          // D_EXP / HEADS
#define BATCH   2
#define SEQ     2048
#define ROWS    (BATCH * SEQ)        // 4096 token rows
#define QKVP_N  (4 * D_EXP)          // 8192

typedef __attribute__((ext_vector_type(16))) __bf16 v16bf;
typedef __attribute__((ext_vector_type(8)))  float  v8f;

union V8 { v8f v; float f[8]; };

// ---------------------------------------------------------------------------
// bf16 helpers (storage as u16; math in fp32)
// ---------------------------------------------------------------------------
__device__ __forceinline__ unsigned short f2bf(float f) {
    unsigned int u = __builtin_bit_cast(unsigned int, f);
    unsigned int lsb = (u >> 16) & 1u;
    u += 0x7FFFu + lsb;                 // round-to-nearest-even
    return (unsigned short)(u >> 16);
}
__device__ __forceinline__ float bf2f(unsigned short h) {
    unsigned int u = ((unsigned int)h) << 16;
    return __builtin_bit_cast(float, u);
}

// ---------------------------------------------------------------------------
// WMMA fragment loaders (CDNA5 16x16x32 bf16 layouts, wave32)
//  A (16xK):  lane m = lane%16 ; elems[0..7]=K[k0+half*8 ..], elems[8..15]=K[k0+16+half*8 ..]
//  B (Kx16):  lane n = lane%16 ; elems[0..7]=K[k0+half*8 ..], elems[8..15]=K[k0+16+half*8 ..]
// ---------------------------------------------------------------------------
union FragCvt { struct { uint4 lo, hi; } u; v16bf v; };

__device__ __forceinline__ v16bf load_a_frag(const unsigned short* base, int stride,
                                             int rowbase, int k0) {
    int lane = threadIdx.x & 31;
    int half = lane >> 4;
    const unsigned short* p =
        base + (size_t)(rowbase + (lane & 15)) * stride + (k0 + half * 8);
    FragCvt f;
    f.u.lo = *(const uint4*)(p);        // K-run 0..7
    f.u.hi = *(const uint4*)(p + 16);   // K-run 16..23 (offset +16 elems)
    return f.v;
}

__device__ __forceinline__ v16bf load_b_frag(const unsigned short* base, int stride,
                                             int colbase, int k0) {
    int lane = threadIdx.x & 31;
    int half = lane >> 4;
    const unsigned short* p =
        base + (size_t)(colbase + (lane & 15)) * stride + (k0 + half * 8);
    FragCvt f;
    f.u.lo = *(const uint4*)(p);
    f.u.hi = *(const uint4*)(p + 16);   // K-run 16..23 (offset +16 elems)
    return f.v;
}

__device__ __forceinline__ v8f wmma_bf16(v16bf a, v16bf b, v8f c) {
    return __builtin_amdgcn_wmma_f32_16x16x32_bf16(false, a, false, b,
                                                   (short)0, c, false, false);
}

// ---------------------------------------------------------------------------
// Block reduction (256 threads)
// ---------------------------------------------------------------------------
__device__ __forceinline__ float block_sum(float v, float* red) {
    int tid = threadIdx.x;
    red[tid] = v;
    __syncthreads();
    for (int s = 128; s > 0; s >>= 1) {
        if (tid < s) red[tid] += red[tid + s];
        __syncthreads();
    }
    float r = red[0];
    __syncthreads();
    return r;
}

// ---------------------------------------------------------------------------
// Kernel: fp32 -> bf16 convert (weights)
// ---------------------------------------------------------------------------
__global__ __launch_bounds__(256) void cvt_bf16_kernel(const float* __restrict__ src,
                                                       unsigned short* __restrict__ dst,
                                                       int n) {
    int i = blockIdx.x * 256 + threadIdx.x;
    if (i < n) dst[i] = f2bf(src[i]);
}

// ---------------------------------------------------------------------------
// Kernel: LayerNorm row -> bf16 (input LN) : one block per token row
// ---------------------------------------------------------------------------
__global__ __launch_bounds__(256) void ln_to_bf16_kernel(const float* __restrict__ x,
                                                         const float* __restrict__ g,
                                                         const float* __restrict__ bb,
                                                         unsigned short* __restrict__ out) {
    __shared__ float red[256];
    int row = blockIdx.x, tid = threadIdx.x;
    const float* xr = x + (size_t)row * D_MODEL;
    float s = 0.f;
    for (int i = tid; i < D_MODEL; i += 256) s += xr[i];
    float mu = block_sum(s, red) * (1.0f / D_MODEL);
    float var = 0.f;
    for (int i = tid; i < D_MODEL; i += 256) { float t = xr[i] - mu; var += t * t; }
    float rstd = rsqrtf(block_sum(var, red) * (1.0f / D_MODEL) + 1e-5f);
    unsigned short* orow = out + (size_t)row * D_MODEL;
    for (int i = tid; i < D_MODEL; i += 256)
        orow[i] = f2bf((xr[i] - mu) * rstd * g[i] + bb[i]);
}

// ---------------------------------------------------------------------------
// Kernel: LayerNorm row -> fp32 (output LN)
// ---------------------------------------------------------------------------
__global__ __launch_bounds__(256) void ln_to_f32_kernel(const float* __restrict__ x,
                                                        const float* __restrict__ g,
                                                        const float* __restrict__ bb,
                                                        float* __restrict__ out) {
    __shared__ float red[256];
    int row = blockIdx.x, tid = threadIdx.x;
    const float* xr = x + (size_t)row * D_MODEL;
    float s = 0.f;
    for (int i = tid; i < D_MODEL; i += 256) s += xr[i];
    float mu = block_sum(s, red) * (1.0f / D_MODEL);
    float var = 0.f;
    for (int i = tid; i < D_MODEL; i += 256) { float t = xr[i] - mu; var += t * t; }
    float rstd = rsqrtf(block_sum(var, red) * (1.0f / D_MODEL) + 1e-5f);
    float* orow = out + (size_t)row * D_MODEL;
    for (int i = tid; i < D_MODEL; i += 256)
        orow[i] = (xr[i] - mu) * rstd * g[i] + bb[i];
}

// ---------------------------------------------------------------------------
// Kernel: GEMM1  qkvp = h(4096x1024) @ w_in^T(1024x8192), bf16 WMMA, f32 acc.
// Register-blocked: one wave computes a 64x64 macro-tile (4x4 WMMA tiles),
// so 8 fragment loads feed 16 WMMAs per K-step (1:1 wmma : b128-load).
// grid = (128 n-macro, 8), block = 256 (8 waves -> 8 m-macro tiles)
// Epilogue scatters into q / k / vT / p layouts.
// ---------------------------------------------------------------------------
__global__ __launch_bounds__(256) void gemm_qkvp_kernel(
        const unsigned short* __restrict__ A,      // h_bf   4096 x 1024
        const unsigned short* __restrict__ W,      // w_in_bf 8192 x 1024
        unsigned short* __restrict__ qh,           // [b][h][i][d] bf16
        unsigned short* __restrict__ kh,           // [b][h][i][d] bf16 (pre-smear)
        unsigned short* __restrict__ vT,           // [b][h][d][i] bf16
        float* __restrict__ pbuf) {                // [row][h*128+d] f32
    int wave = threadIdx.x >> 5;
    int lane = threadIdx.x & 31;
    int n0 = blockIdx.x * 64;
    int m0 = (blockIdx.y * 8 + wave) * 64;

    V8 acc[4][4];
#pragma unroll
    for (int i = 0; i < 4; ++i)
#pragma unroll
        for (int j = 0; j < 4; ++j)
            acc[i][j].v = (v8f){0.f,0.f,0.f,0.f,0.f,0.f,0.f,0.f};

    for (int k0 = 0; k0 < D_MODEL; k0 += 32) {
        v16bf af[4], bf[4];
#pragma unroll
        for (int i = 0; i < 4; ++i) af[i] = load_a_frag(A, D_MODEL, m0 + 16 * i, k0);
#pragma unroll
        for (int j = 0; j < 4; ++j) bf[j] = load_b_frag(W, D_MODEL, n0 + 16 * j, k0);
#pragma unroll
        for (int i = 0; i < 4; ++i)
#pragma unroll
            for (int j = 0; j < 4; ++j)
                acc[i][j].v = wmma_bf16(af[i], bf[j], acc[i][j].v);
    }

    int half = lane >> 4, nn = lane & 15;
#pragma unroll
    for (int j = 0; j < 4; ++j) {
        int gn  = n0 + 16 * j + nn;
        int seg = gn >> 11;                 // 0:q 1:k 2:v 3:p
        int loc = gn & (D_EXP - 1);
        int hh  = loc >> 7, dd = loc & (D_HEAD - 1);
#pragma unroll
        for (int i = 0; i < 4; ++i) {
#pragma unroll
            for (int r = 0; r < 8; ++r) {
                int gm = m0 + 16 * i + half * 8 + r;
                int b  = gm >> 11, ii = gm & (SEQ - 1);
                float val = acc[i][j].f[r];
                size_t head_base = (size_t)(b * HEADS + hh);
                if (seg == 0)
                    qh[(head_base * SEQ + ii) * D_HEAD + dd] = f2bf(val);
                else if (seg == 1)
                    kh[(head_base * SEQ + ii) * D_HEAD + dd] = f2bf(val);
                else if (seg == 2)
                    vT[(head_base * D_HEAD + dd) * SEQ + ii] = f2bf(val);
                else
                    pbuf[(size_t)gm * D_EXP + loc] = val;
            }
        }
    }
}

// ---------------------------------------------------------------------------
// Kernel: key smearing  k_i <- (1-s)k_i + s k_{i-1},  s = sigmoid(smear[h])
// ---------------------------------------------------------------------------
__global__ __launch_bounds__(256) void smear_kernel(const unsigned short* __restrict__ k0h,
                                                    unsigned short* __restrict__ ksm,
                                                    const float* __restrict__ smf) {
    size_t idx = (size_t)blockIdx.x * 256 + threadIdx.x;   // over 2*16*2048*128
    size_t t  = idx >> 7;           // (b,h,i)
    int   i   = (int)(t & (SEQ - 1));
    int   h   = (int)((t >> 11) & (HEADS - 1));
    float s   = 1.0f / (1.0f + __expf(-smf[h]));
    float kv  = bf2f(k0h[idx]);
    float kp  = (i == 0) ? 0.0f : bf2f(k0h[idx - D_HEAD]);
    ksm[idx]  = f2bf((1.0f - s) * kv + s * kp);
}

// ---------------------------------------------------------------------------
// Kernel: flash attention. One wave owns TWO 16-row query tiles (32 rows) of
// one (b,h): every K fragment feeds 2 QK^T WMMAs and every V fragment feeds
// 2 PV WMMAs -> 32 WMMAs per 32 b128 loads per 32-key chunk (1:1).
// Online softmax with ALiBi+causal; P tiles go through per-wave LDS
// (D-layout -> A-layout).
// grid = 256 blocks x 256 threads (8 waves) = 2048 macro-tiles.
// ---------------------------------------------------------------------------
__global__ __launch_bounds__(256) void attn_kernel(
        const unsigned short* __restrict__ qh,
        const unsigned short* __restrict__ kh,      // smeared keys
        const unsigned short* __restrict__ vT,
        const float* __restrict__ slopes,
        float* __restrict__ obuf) {                 // [row][h*128+d] f32
    __shared__ __align__(16) unsigned short plds_all[8][2][16 * 32];   // 16 KB
    int wave = threadIdx.x >> 5;
    int lane = threadIdx.x & 31;

    int wid = blockIdx.x * 8 + wave;        // 0..2047
    int it  = wid & 63;                     // 64 macro-tiles of 32 rows
    int h   = (wid >> 6) & (HEADS - 1);
    int b   = wid >> 10;
    int i0  = it << 5;                      // 32 query rows per wave

    size_t hb = (size_t)(b * HEADS + h);
    const unsigned short* qb = qh + hb * SEQ * D_HEAD;
    const unsigned short* kb = kh + hb * SEQ * D_HEAD;
    const unsigned short* vb = vT + hb * D_HEAD * SEQ;
    float slope = slopes[h];

    v16bf qf[2][4];
#pragma unroll
    for (int u = 0; u < 2; ++u)
#pragma unroll
        for (int c = 0; c < 4; ++c)
            qf[u][c] = load_a_frag(qb, D_HEAD, i0 + u * 16, c * 32);

    V8 oa[2][8];
#pragma unroll
    for (int u = 0; u < 2; ++u)
#pragma unroll
        for (int t = 0; t < 8; ++t)
            oa[u][t].v = (v8f){0.f,0.f,0.f,0.f,0.f,0.f,0.f,0.f};
    float mrow[2][8], lrow[2][8];
#pragma unroll
    for (int u = 0; u < 2; ++u)
#pragma unroll
        for (int r = 0; r < 8; ++r) { mrow[u][r] = -1e30f; lrow[u][r] = 0.0f; }

    int half = lane >> 4, nn = lane & 15;
    const float scale = 0.08838834764831845f;       // 1/sqrt(128)
    int nkeys = i0 + 32;                            // multiple of 32

    for (int J = 0; J < nkeys; J += 32) {
        // ---- S = Q K^T : K fragments shared across both query tiles ----
        V8 s[2][2];
#pragma unroll
        for (int u = 0; u < 2; ++u)
#pragma unroll
            for (int t = 0; t < 2; ++t)
                s[u][t].v = (v8f){0.f,0.f,0.f,0.f,0.f,0.f,0.f,0.f};
#pragma unroll
        for (int c = 0; c < 4; ++c) {
            v16bf b0 = load_b_frag(kb, D_HEAD, J,      c * 32);
            v16bf b1 = load_b_frag(kb, D_HEAD, J + 16, c * 32);
            s[0][0].v = wmma_bf16(qf[0][c], b0, s[0][0].v);
            s[0][1].v = wmma_bf16(qf[0][c], b1, s[0][1].v);
            s[1][0].v = wmma_bf16(qf[1][c], b0, s[1][0].v);
            s[1][1].v = wmma_bf16(qf[1][c], b1, s[1][1].v);
        }

        // ---- online softmax per query tile ----
        int   j0  = J + nn,        j1  = J + 16 + nn;
        float ab0 = slope * (float)j0, ab1 = slope * (float)j1;
#pragma unroll
        for (int u = 0; u < 2; ++u) {
            unsigned short* plds = plds_all[wave][u];
#pragma unroll
            for (int r = 0; r < 8; ++r) {
                int irow = i0 + u * 16 + half * 8 + r;
                float a0 = (j0 <= irow) ? s[u][0].f[r] * scale + ab0 : -1e30f;
                float a1 = (j1 <= irow) ? s[u][1].f[r] * scale + ab1 : -1e30f;
                // row-max across the 16 lanes of this half
                float mx = fmaxf(a0, a1);
                for (int off = 8; off > 0; off >>= 1)
                    mx = fmaxf(mx, __shfl_xor(mx, off, 32));
                float nm = fmaxf(mrow[u][r], mx);
                float p0 = __expf(a0 - nm), p1 = __expf(a1 - nm);
                float ps = p0 + p1;
                for (int off = 8; off > 0; off >>= 1)
                    ps += __shfl_xor(ps, off, 32);
                float corr = __expf(mrow[u][r] - nm);
                mrow[u][r] = nm;
                lrow[u][r] = lrow[u][r] * corr + ps;
#pragma unroll
                for (int t = 0; t < 8; ++t) oa[u][t].f[r] *= corr;
                int prow = half * 8 + r;
                plds[prow * 32 + nn]      = f2bf(p0);
                plds[prow * 32 + 16 + nn] = f2bf(p1);
            }
        }

        asm volatile("s_wait_dscnt 0" ::: "memory");   // LDS writes visible to our reads
        v16bf pa0 = load_a_frag(plds_all[wave][0], 32, 0, 0);
        v16bf pa1 = load_a_frag(plds_all[wave][1], 32, 0, 0);

        // ---- O += P V : V fragments shared across both query tiles ----
#pragma unroll
        for (int t = 0; t < 8; ++t) {
            v16bf bv = load_b_frag(vb, SEQ, t * 16, J);   // V^T rows d, contiguous j
            oa[0][t].v = wmma_bf16(pa0, bv, oa[0][t].v);
            oa[1][t].v = wmma_bf16(pa1, bv, oa[1][t].v);
        }
    }

    // normalize + store O as fp32 [row][h*128 + d]
#pragma unroll
    for (int u = 0; u < 2; ++u)
#pragma unroll
        for (int r = 0; r < 8; ++r) {
            float inv = 1.0f / lrow[u][r];
            int gm = b * SEQ + i0 + u * 16 + half * 8 + r;
            float* orow = obuf + (size_t)gm * D_EXP + h * D_HEAD + nn;
#pragma unroll
            for (int t = 0; t < 8; ++t)
                orow[t * 16] = oa[u][t].f[r] * inv;
        }
}

// ---------------------------------------------------------------------------
// Kernel: gating  g = bf16( silu(p) * o )
// ---------------------------------------------------------------------------
__global__ __launch_bounds__(256) void gate_kernel(const float* __restrict__ p,
                                                   const float* __restrict__ o,
                                                   unsigned short* __restrict__ g) {
    size_t idx = (size_t)blockIdx.x * 256 + threadIdx.x;   // 4096*2048
    float pv  = p[idx];
    float sil = pv / (1.0f + __expf(-pv));
    g[idx] = f2bf(sil * o[idx]);
}

// ---------------------------------------------------------------------------
// Kernel: GEMM2  y(4096x1024) = g(4096x2048) @ w_out^T, bf16 WMMA, f32 out.
// Register-blocked 64x64 macro-tile per wave (4x4 WMMA tiles).
// grid = (16 n-macro, 8), block = 256 (8 waves)
// ---------------------------------------------------------------------------
__global__ __launch_bounds__(256) void gemm_out_kernel(
        const unsigned short* __restrict__ G,      // 4096 x 2048 bf16
        const unsigned short* __restrict__ Wo,     // 1024 x 2048 bf16
        float* __restrict__ y) {                   // 4096 x 1024 f32
    int wave = threadIdx.x >> 5;
    int lane = threadIdx.x & 31;
    int n0 = blockIdx.x * 64;
    int m0 = (blockIdx.y * 8 + wave) * 64;

    V8 acc[4][4];
#pragma unroll
    for (int i = 0; i < 4; ++i)
#pragma unroll
        for (int j = 0; j < 4; ++j)
            acc[i][j].v = (v8f){0.f,0.f,0.f,0.f,0.f,0.f,0.f,0.f};

    for (int k0 = 0; k0 < D_EXP; k0 += 32) {
        v16bf af[4], bf[4];
#pragma unroll
        for (int i = 0; i < 4; ++i) af[i] = load_a_frag(G,  D_EXP, m0 + 16 * i, k0);
#pragma unroll
        for (int j = 0; j < 4; ++j) bf[j] = load_b_frag(Wo, D_EXP, n0 + 16 * j, k0);
#pragma unroll
        for (int i = 0; i < 4; ++i)
#pragma unroll
            for (int j = 0; j < 4; ++j)
                acc[i][j].v = wmma_bf16(af[i], bf[j], acc[i][j].v);
    }

    int half = lane >> 4, nn = lane & 15;
#pragma unroll
    for (int i = 0; i < 4; ++i)
#pragma unroll
        for (int j = 0; j < 4; ++j)
#pragma unroll
            for (int r = 0; r < 8; ++r) {
                int gm = m0 + 16 * i + half * 8 + r;
                y[(size_t)gm * D_MODEL + n0 + 16 * j + nn] = acc[i][j].f[r];
            }
}

// ---------------------------------------------------------------------------
// Host-side launch
// ---------------------------------------------------------------------------
extern "C" void kernel_launch(void* const* d_in, const int* in_sizes, int n_in,
                              void* d_out, int out_size, void* d_ws, size_t ws_size,
                              hipStream_t stream) {
    const float* x        = (const float*)d_in[0];
    const float* w_in     = (const float*)d_in[1];
    const float* w_out    = (const float*)d_in[2];
    const float* ln_in_g  = (const float*)d_in[3];
    const float* ln_in_b  = (const float*)d_in[4];
    const float* ln_out_g = (const float*)d_in[5];
    const float* ln_out_b = (const float*)d_in[6];
    const float* slopes   = (const float*)d_in[7];
    const float* smearf   = (const float*)d_in[8];
    float* out = (float*)d_out;

    char* ws = (char*)d_ws;
    size_t off = 0;
    auto take = [&](size_t bytes) { size_t o = off; off += (bytes + 255) & ~(size_t)255; return o; };

    unsigned short* h_bf   = (unsigned short*)(ws + take((size_t)ROWS  * D_MODEL * 2)); //  8 MB
    unsigned short* win_bf = (unsigned short*)(ws + take((size_t)QKVP_N * D_MODEL * 2)); // 16 MB
    unsigned short* wout_bf= (unsigned short*)(ws + take((size_t)D_MODEL * D_EXP * 2));  //  4 MB
    unsigned short* qh     = (unsigned short*)(ws + take((size_t)ROWS * D_EXP * 2));     // 16 MB
    unsigned short* kh     = (unsigned short*)(ws + take((size_t)ROWS * D_EXP * 2));     // 16 MB
    unsigned short* ksm    = (unsigned short*)(ws + take((size_t)ROWS * D_EXP * 2));     // 16 MB
    unsigned short* vTb    = (unsigned short*)(ws + take((size_t)ROWS * D_EXP * 2));     // 16 MB
    float*          pbuf   = (float*)         (ws + take((size_t)ROWS * D_EXP * 4));     // 32 MB
    float*          obuf   = (float*)         (ws + take((size_t)ROWS * D_EXP * 4));     // 32 MB
    unsigned short* gbuf   = (unsigned short*)(ws + take((size_t)ROWS * D_EXP * 2));     // 16 MB
    float*          ybuf   = (float*)         (ws + take((size_t)ROWS * D_MODEL * 4));   // 16 MB
    (void)ws_size; (void)in_sizes; (void)n_in; (void)out_size;

    // 1. weight conversion to bf16
    cvt_bf16_kernel<<<(QKVP_N * D_MODEL) / 256, 256, 0, stream>>>(w_in,  win_bf,  QKVP_N * D_MODEL);
    cvt_bf16_kernel<<<(D_MODEL * D_EXP) / 256, 256, 0, stream>>>(w_out, wout_bf, D_MODEL * D_EXP);

    // 2. input LayerNorm -> bf16 activations
    ln_to_bf16_kernel<<<ROWS, 256, 0, stream>>>(x, ln_in_g, ln_in_b, h_bf);

    // 3. fused QKVP GEMM (WMMA, 64x64 register-blocked) with scatter epilogue
    gemm_qkvp_kernel<<<dim3(QKVP_N / 64, (ROWS / 64) / 8), 256, 0, stream>>>(
        h_bf, win_bf, qh, kh, vTb, pbuf);

    // 4. key smearing
    smear_kernel<<<(ROWS * D_EXP) / 256, 256, 0, stream>>>(kh, ksm, smearf);

    // 5. flash attention (WMMA QK^T and PV, 32 query rows per wave)
    attn_kernel<<<(BATCH * HEADS * (SEQ / 32)) / 8, 256, 0, stream>>>(
        qh, ksm, vTb, slopes, obuf);

    // 6. SiLU gating
    gate_kernel<<<(ROWS * D_EXP) / 256, 256, 0, stream>>>(pbuf, obuf, gbuf);

    // 7. output GEMM (WMMA, 64x64 register-blocked)
    gemm_out_kernel<<<dim3(D_MODEL / 64, (ROWS / 64) / 8), 256, 0, stream>>>(
        gbuf, wout_bf, ybuf);

    // 8. output LayerNorm -> d_out
    ln_to_f32_kernel<<<ROWS, 256, 0, stream>>>(ybuf, ln_out_g, ln_out_b, out);
}
